// Decoder_12524124635694
// MI455X (gfx1250) — matrix operand order repaired and checked by reading, not compiled, not run on hardware
//
#include <hip/hip_runtime.h>
#include <hip/hip_bf16.h>

typedef __attribute__((ext_vector_type(16))) _Float16 v16h;
typedef __attribute__((ext_vector_type(8)))  float    v8f;
typedef __attribute__((ext_vector_type(4)))  float    v4f;

#define EPS 1e-5f
#define STATS_BLOCKS 512

// ---------------------------------------------------------------------------
// Kernel 1a: per-block partial column sums / sums-of-squares of G = S[idx].
// ---------------------------------------------------------------------------
__global__ __launch_bounds__(256) void stats_kernel(
    const float* __restrict__ S, const int* __restrict__ idx,
    int Nf, int C, float* __restrict__ partials) {
  const int tid      = threadIdx.x;
  const int cthreads = C >> 2;            // 32 / 64 / 128
  const int nrg      = 256 / cthreads;    // 8 / 4 / 2
  const int ct       = tid % cthreads;
  const int rg       = tid / cthreads;
  const int c        = ct * 4;

  float4 a = {0.f, 0.f, 0.f, 0.f};
  float4 q = {0.f, 0.f, 0.f, 0.f};
  for (int r = blockIdx.x * nrg + rg; r < Nf; r += gridDim.x * nrg) {
    const float4 v = *(const float4*)(S + (long)idx[r] * C + c);
    a.x += v.x; a.y += v.y; a.z += v.z; a.w += v.w;
    q.x += v.x * v.x; q.y += v.y * v.y; q.z += v.z * v.z; q.w += v.w * v.w;
  }

  __shared__ float lsum[1024];
  __shared__ float lsq[1024];
  const int slot = tid * 4;
  *(float4*)&lsum[slot] = a;
  *(float4*)&lsq[slot]  = q;
  __syncthreads();
  for (int s = nrg >> 1; s > 0; s >>= 1) {
    if (rg < s) {
      int other = slot + s * cthreads * 4;
#pragma unroll
      for (int i = 0; i < 4; ++i) {
        lsum[slot + i] += lsum[other + i];
        lsq[slot + i]  += lsq[other + i];
      }
    }
    __syncthreads();
  }
  if (rg == 0) {
    float* dst = partials + (size_t)blockIdx.x * 2 * C;
    *(float4*)&dst[c]     = *(float4*)&lsum[slot];
    *(float4*)&dst[C + c] = *(float4*)&lsq[slot];
  }
}

// Kernel 1b: reduce block partials -> sums[0..C-1]=sum, sums[C..2C-1]=sumsq
__global__ __launch_bounds__(256) void reduce_partials(
    const float* __restrict__ partials, int twoC, float* __restrict__ sums) {
  int j = blockIdx.x * 256 + threadIdx.x;
  if (j >= twoC) return;
  float a = 0.f;
  for (int b = 0; b < STATS_BLOCKS; ++b) a += partials[(size_t)b * twoC + j];
  sums[j] = a;
}

// ---------------------------------------------------------------------------
// Kernel 2: fold BN into weights, emitting Wfrag in WMMA B-fragment-major
// layout; bias[o] = sum_c (bn_b_c - mean_c*s_c) * W[o][c] in fp32.
// ---------------------------------------------------------------------------
__global__ __launch_bounds__(256) void fold_kernel(
    const float* __restrict__ W, const float* __restrict__ bn_w,
    const float* __restrict__ bn_b, const float* __restrict__ sums,
    float invN, int C, int COUT,
    _Float16* __restrict__ Wfrag, float* __restrict__ bias) {
  const int o     = blockIdx.x;
  const int NT    = COUT >> 4;
  const int nt    = o >> 4;
  const int laneN = o & 15;
  float part = 0.f;
  for (int c = threadIdx.x; c < C; c += 256) {
    float mean = sums[c] * invN;
    float var  = sums[C + c] * invN - mean * mean;
    float s    = bn_w[c] * rsqrtf(var + EPS);
    float w    = W[(long)o * C + c];
    int kt   = c >> 5;
    int kk   = c & 31;
    int lane = (kk & 16) + laneN;   // lanes 16-31 hold K-range [16,32)
    int h    = kk & 15;
    Wfrag[(((size_t)kt * NT + nt) * 32 + lane) * 16 + h] = (_Float16)(w * s);
    part += (bn_b[c] - mean * s) * w;
  }
  __shared__ float red[256];
  red[threadIdx.x] = part;
  __syncthreads();
  for (int off = 128; off > 0; off >>= 1) {
    if (threadIdx.x < off) red[threadIdx.x] += red[threadIdx.x + off];
    __syncthreads();
  }
  if (threadIdx.x == 0) bias[o] = red[0];
}

// ---------------------------------------------------------------------------
// Kernel 3: gather-fused WMMA GEMM + bias + residual (double-buffered LDS).
// Steady-state K-step issue order: B frags -> next-A gmem -> A frags (LDS) ->
// WMMA x4 -> cvt+store next tile -> barrier.  B/LDS are the only WMMA waits;
// the random-row gather completes under the WMMAs.
// A base pointers stay loop-invariant (scalar koff advances) so the gather
// lowers to GLOBAL (not FLAT) loads and stays off the DScnt path.
// ---------------------------------------------------------------------------
#define BM 128
#define BN 64
#define AS_STRIDE 40   // halves: 80B rows -> 16B-aligned frags, conflict-free

__global__ __launch_bounds__(256) void gemm_kernel(
    const float* __restrict__ S, const int* __restrict__ idx,
    const _Float16* __restrict__ Wfrag, const float* __restrict__ bias,
    const float* __restrict__ res, float* __restrict__ out,
    int C, int COUT) {
  __shared__ _Float16 As[2][BM * AS_STRIDE];

  const int tid  = threadIdx.x;
  const int lane = tid & 31;
  const int w    = tid >> 5;
  const int wr   = w & 3;     // rows [wr*32, wr*32+32)
  const int wc   = w >> 2;    // cols [wc*32, wc*32+32)

  const long rowBase = (long)blockIdx.x * BM;
  const int  colBase = blockIdx.y * BN;
  const int  NT      = COUT >> 4;
  const int  nt0     = (colBase >> 4) + wc * 2;

  // A-tile loading: 8 threads/row (float4 each = 32 ch), 4 passes of 32 rows.
  const int aRow = tid >> 3;
  const int aCol = (tid & 7) * 4;

  // Loop-invariant gather base pointers (idx loads hoisted).
  const float* ap0 = S + (long)idx[rowBase +  0 + aRow] * C + aCol;
  const float* ap1 = S + (long)idx[rowBase + 32 + aRow] * C + aCol;
  const float* ap2 = S + (long)idx[rowBase + 64 + aRow] * C + aCol;
  const float* ap3 = S + (long)idx[rowBase + 96 + aRow] * C + aCol;

  // B fragment pointers (fragment-major), advanced one K-tile per step.
  const size_t    bstep = (size_t)NT * 32 * 16;
  const _Float16* bp0   = Wfrag + ((size_t)nt0 * 32 + lane) * 16;
  const _Float16* bp1   = bp0 + 32 * 16;

  v8f acc00 = {}, acc01 = {}, acc10 = {}, acc11 = {};

  const int m    = lane & 15;
  const int asel = (lane < 16) ? 0 : 16;  // A fragment K base (halves)
  const _Float16* arBase  = &As[0][(wr * 32 + m) * AS_STRIDE + asel];
  const int bufHalves     = BM * AS_STRIDE;

  auto loadB = [&](int t, v16h& b0, v16h& b1) {
    const v4f* q0 = (const v4f*)(bp0 + (size_t)t * bstep);
    ((v4f*)&b0)[0] = q0[0];
    ((v4f*)&b0)[1] = q0[1];
    const v4f* q1 = (const v4f*)(bp1 + (size_t)t * bstep);
    ((v4f*)&b1)[0] = q1[0];
    ((v4f*)&b1)[1] = q1[1];
  };
  auto loadA = [&](int cur, v16h& a0, v16h& a1) {
    const _Float16* ar0 = arBase + cur * bufHalves;
    ((v4f*)&a0)[0] = *(const v4f*)ar0;
    ((v4f*)&a0)[1] = *(const v4f*)(ar0 + 8);
    const _Float16* ar1 = ar0 + 16 * AS_STRIDE;
    ((v4f*)&a1)[0] = *(const v4f*)ar1;
    ((v4f*)&a1)[1] = *(const v4f*)(ar1 + 8);
  };
  auto gatherA = [&](int koff, float4* va) {
    va[0] = *(const float4*)(ap0 + koff);
    va[1] = *(const float4*)(ap1 + koff);
    va[2] = *(const float4*)(ap2 + koff);
    va[3] = *(const float4*)(ap3 + koff);
  };
  auto stageA = [&](int buf, const float4* va) {
    _Float16* dst = &As[buf][aRow * AS_STRIDE + aCol];
#pragma unroll
    for (int p = 0; p < 4; ++p) {
      union { _Float16 h[4]; uint2 u; } pk;
      pk.h[0] = (_Float16)va[p].x; pk.h[1] = (_Float16)va[p].y;
      pk.h[2] = (_Float16)va[p].z; pk.h[3] = (_Float16)va[p].w;
      *(uint2*)(dst + p * 32 * AS_STRIDE) = pk.u;
    }
  };

  // ---- prologue: stage tile 0 into buffer 0 ----
  {
    float4 va[4];
    gatherA(0, va);
    stageA(0, va);
  }
  __syncthreads();

  const int T = C >> 5;
  int t = 0;
  // ---- steady state: branch-free body, last step peeled ----
  for (; t < T - 1; ++t) {
    const int cur = t & 1;

    v16h b0, b1;
    loadB(t, b0, b1);                 // issued first: WMMA waits only on these

    float4 va[4];
    gatherA((t + 1) * 32, va);        // next tile, completes under the WMMAs

    v16h a0, a1;
    loadA(cur, a0, a1);

    acc00 = __builtin_amdgcn_wmma_f32_16x16x32_f16(false, a0, false, b0, (short)0, acc00, false, false);
    acc01 = __builtin_amdgcn_wmma_f32_16x16x32_f16(false, a0, false, b1, (short)0, acc01, false, false);
    acc10 = __builtin_amdgcn_wmma_f32_16x16x32_f16(false, a1, false, b0, (short)0, acc10, false, false);
    acc11 = __builtin_amdgcn_wmma_f32_16x16x32_f16(false, a1, false, b1, (short)0, acc11, false, false);

    stageA(cur ^ 1, va);

    __syncthreads();
  }
  // ---- final K-step: no next tile ----
  {
    const int cur = t & 1;
    v16h b0, b1, a0, a1;
    loadB(t, b0, b1);
    loadA(cur, a0, a1);
    acc00 = __builtin_amdgcn_wmma_f32_16x16x32_f16(false, a0, false, b0, (short)0, acc00, false, false);
    acc01 = __builtin_amdgcn_wmma_f32_16x16x32_f16(false, a0, false, b1, (short)0, acc01, false, false);
    acc10 = __builtin_amdgcn_wmma_f32_16x16x32_f16(false, a1, false, b0, (short)0, acc10, false, false);
    acc11 = __builtin_amdgcn_wmma_f32_16x16x32_f16(false, a1, false, b1, (short)0, acc11, false, false);
  }

  // --- epilogue: bias + residual, fp32 ---
  const int n    = lane & 15;
  const int mOff = (lane < 16) ? 0 : 8;
#pragma unroll
  for (int tr = 0; tr < 2; ++tr) {
#pragma unroll
    for (int tc = 0; tc < 2; ++tc) {
      v8f acc = (tr == 0) ? ((tc == 0) ? acc00 : acc01)
                          : ((tc == 0) ? acc10 : acc11);
      long rg0 = rowBase + wr * 32 + tr * 16 + mOff;
      int  cg  = colBase + wc * 32 + tc * 16 + n;
      float bv = bias[cg];
#pragma unroll
      for (int v = 0; v < 8; ++v) {
        long rg = rg0 + v;
        out[rg * COUT + cg] = res[rg * COUT + cg] + bv + acc[v];
      }
    }
  }
}

// ---------------------------------------------------------------------------
// Launch: 3 chained stages (coarse -> fine), all on `stream`.
// ---------------------------------------------------------------------------
extern "C" void kernel_launch(void* const* d_in, const int* in_sizes, int n_in,
                              void* d_out, int out_size, void* d_ws, size_t ws_size,
                              hipStream_t stream) {
  (void)in_sizes; (void)n_in; (void)out_size; (void)ws_size;

  const float* f0   = (const float*)d_in[0];   // [262144,64]
  const float* f1   = (const float*)d_in[1];   // [65536,128]
  const float* f2   = (const float*)d_in[2];   // [16384,256]
  const float* f3   = (const float*)d_in[3];   // [4096,512]
  const float* bnw2 = (const float*)d_in[4];
  const float* bnb2 = (const float*)d_in[5];
  const float* W2   = (const float*)d_in[6];   // [256,512]
  const float* bnw1 = (const float*)d_in[7];
  const float* bnb1 = (const float*)d_in[8];
  const float* W1   = (const float*)d_in[9];   // [128,256]
  const float* bnw0 = (const float*)d_in[10];
  const float* bnb0 = (const float*)d_in[11];
  const float* W0   = (const float*)d_in[12];  // [64,128]
  const int* idx0   = (const int*)d_in[13];    // [16384]  f3 -> level2
  const int* idx1   = (const int*)d_in[14];    // [65536]  f2 -> level1
  const int* idx2   = (const int*)d_in[15];    // [262144] f1 -> level0

  const int N0 = 262144, N1 = 65536, N2 = 16384;

  // workspace layout (all slices 256B-aligned)
  char* ws = (char*)d_ws;
  float*    f2p   = (float*)ws;   ws += (size_t)N2 * 256 * 4;            // 16 MB
  float*    f1p   = (float*)ws;   ws += (size_t)N1 * 128 * 4;            // 32 MB
  float*    part  = (float*)ws;   ws += (size_t)STATS_BLOCKS * 1024 * 4; // 2 MB
  float*    sums  = (float*)ws;   ws += 2 * 512 * 4;                     // 4 KB
  float*    bias  = (float*)ws;   ws += 256 * 4;                         // 1 KB
  _Float16* Wf    = (_Float16*)ws; ws += (size_t)512 * 256 * 2;          // 256 KB

  dim3 blk(256);

  // ---- Stage A: f2' = f2 + decode(f3[idx0]); 512 -> 256 on N2 rows ----
  stats_kernel<<<STATS_BLOCKS, blk, 0, stream>>>(f3, idx0, N2, 512, part);
  reduce_partials<<<4, blk, 0, stream>>>(part, 2 * 512, sums);
  fold_kernel<<<256, blk, 0, stream>>>(W2, bnw2, bnb2, sums, 1.0f / N2, 512, 256, Wf, bias);
  gemm_kernel<<<dim3(N2 / BM, 256 / BN), blk, 0, stream>>>(f3, idx0, Wf, bias, f2, f2p, 512, 256);

  // ---- Stage B: f1' = f1 + decode(f2'[idx1]); 256 -> 128 on N1 rows ----
  stats_kernel<<<STATS_BLOCKS, blk, 0, stream>>>(f2p, idx1, N1, 256, part);
  reduce_partials<<<2, blk, 0, stream>>>(part, 2 * 256, sums);
  fold_kernel<<<128, blk, 0, stream>>>(W1, bnw1, bnb1, sums, 1.0f / N1, 256, 128, Wf, bias);
  gemm_kernel<<<dim3(N1 / BM, 128 / BN), blk, 0, stream>>>(f2p, idx1, Wf, bias, f1, f1p, 256, 128);

  // ---- Stage C: f0' = f0 + decode(f1'[idx2]); 128 -> 64 on N0 rows ----
  stats_kernel<<<STATS_BLOCKS, blk, 0, stream>>>(f1p, idx2, N0, 128, part);
  reduce_partials<<<1, blk, 0, stream>>>(part, 2 * 128, sums);
  fold_kernel<<<64, blk, 0, stream>>>(W0, bnw0, bnb0, sums, 1.0f / N0, 128, 64, Wf, bias);
  gemm_kernel<<<dim3(N0 / BM, 64 / BN), blk, 0, stream>>>(f1p, idx2, Wf, bias, f0, (float*)d_out, 128, 64);
}
